// MinkResNet_26225070310078
// MI455X (gfx1250) — compile-verified, matrix operand order checked by reading, not run on hardware
//
#include <hip/hip_runtime.h>
#include <hip/hip_bf16.h>

// ---------------------------------------------------------------------------
// Types / WMMA helpers (CDNA5 / gfx1250, wave32)
// ---------------------------------------------------------------------------
typedef _Float16 half8   __attribute__((ext_vector_type(8)));
typedef _Float16 half16  __attribute__((ext_vector_type(16)));
typedef float    float8  __attribute__((ext_vector_type(8)));
typedef unsigned int u32x4 __attribute__((ext_vector_type(4)));
typedef int      i32x4  __attribute__((ext_vector_type(4)));
typedef int      i32x8  __attribute__((ext_vector_type(8)));

#if __has_builtin(__builtin_amdgcn_tensor_load_to_lds) && \
    __has_builtin(__builtin_amdgcn_s_wait_tensorcnt)
#define CDNA5_HAVE_TDM 1
#else
#define CDNA5_HAVE_TDM 0
#endif

static __device__ inline half16 hcat(half8 lo, half8 hi) {
  return __builtin_shufflevector(lo, hi, 0,1,2,3,4,5,6,7,8,9,10,11,12,13,14,15);
}

static __device__ inline float8 wmma16x16x32(half16 a, half16 b, float8 c) {
  // D = A(16x32 f16) * B(32x16 f16) + C(16x16 f32)
  return __builtin_amdgcn_wmma_f32_16x16x32_f16(false, a, false, b, (short)0, c,
                                                false, false);
}

#if CDNA5_HAVE_TDM
// ---------------------------------------------------------------------------
// Tensor Data Mover: issue one (up to 3D) f16 tile load  global -> LDS.
//   address = ga + 2*(x + y*s0 + z*s1), x<t0, y<t1, z<t2 (t2==0 -> 2D)
//   td* are the REMAINING tensor extents -> hardware zero-fills OOB rows.
// D# packing per CDNA5 ISA ch.8 (group0 128b, group1 256b).
// ---------------------------------------------------------------------------
static __device__ inline void tdm_load_f16_tile(unsigned lds_off,
                                                const void* gptr,
                                                int td0, int td1, int td2,
                                                int t0, int t1, int t2,
                                                long long s0, long long s1) {
  unsigned long long ga = (unsigned long long)gptr;
  u32x4 g0;
  g0[0] = 1u;                                    // count=1, user descriptor
  g0[1] = lds_off;                               // LDS byte address
  g0[2] = (unsigned)ga;                          // global addr [31:0]
  g0[3] = (unsigned)((ga >> 32) & 0x01FFFFFFull) | (2u << 30);  // [56:32]|type=2
  i32x8 g1;
  g1[0] = (1 << 16);                             // data_size = 1 -> 2 bytes
  g1[1] = (td0 & 0xFFFF) << 16;                  // tensor_dim0 [15:0]
  g1[2] = ((td0 >> 16) & 0xFFFF) | ((td1 & 0xFFFF) << 16);
  g1[3] = ((td1 >> 16) & 0xFFFF) | ((t0 & 0xFFFF) << 16);   // tile_dim0
  g1[4] = (t1 & 0xFFFF) | ((t2 & 0xFFFF) << 16); // tile_dim1, tile_dim2
  g1[5] = (int)(s0 & 0xFFFFFFFFll);              // tensor_dim0_stride [31:0]
  g1[6] = (int)((s0 >> 32) & 0xFFFF) | (int)((s1 & 0xFFFF) << 16);
  g1[7] = (int)((s1 >> 16) & 0xFFFFFFFFll);      // tensor_dim1_stride [47:16]
  i32x4 g2;                                      // tensor_dim2 (3D), rest 0
  g2[0] = td2; g2[1] = 0; g2[2] = 0; g2[3] = 0;
  i32x4 g3; g3[0] = 0; g3[1] = 0; g3[2] = 0; g3[3] = 0;
#if defined(__clang_major__) && __clang_major__ >= 23
  i32x8 z8; for (int i = 0; i < 8; ++i) z8[i] = 0;
  __builtin_amdgcn_tensor_load_to_lds(g0, g1, g2, g3, z8, 0);
#else
  __builtin_amdgcn_tensor_load_to_lds(g0, g1, g2, g3, 0);
#endif
}
#endif  // CDNA5_HAVE_TDM

// ---------------------------------------------------------------------------
// f32 -> f16 convert
// ---------------------------------------------------------------------------
__global__ __launch_bounds__(256) void f32_to_f16_kernel(
    const float* __restrict__ src, _Float16* __restrict__ dst, int n) {
  int i = blockIdx.x * 256 + threadIdx.x;
  if (i < n) dst[i] = (_Float16)src[i];
}

// ---------------------------------------------------------------------------
// 3D conv as implicit GEMM on WMMA.
//  - KK (1/3/7) templated -> constant-divisor k decomposition
//  - STEM: Cin==3 (constant); otherwise Cin is a power of two (shift/mask)
//  - all spatial dims are powers of two -> shifts/masks only
//  - block = 4 waves over a 32(m) x 64(n) macro tile: wave = (msel, npair),
//    each wave 16x32 -> 2 WMMAs per K-step
//  - A tile (32x32 f16) gathered to LDS by all 128 threads
//  - B tile (32k x 32n f16) DMA'd by the Tensor Data Mover into per-wave LDS,
//    double buffered, overlapped with compute via s_wait_tensorcnt 1
// ---------------------------------------------------------------------------
template <int KK, bool STEM>
__global__ __launch_bounds__(128) void conv3d_wmma_kernel(
    const _Float16* __restrict__ act, const _Float16* __restrict__ wgt,
    float* __restrict__ out,
    int Di, int Hi, int Wi, int Cin, int cshift,
    int lWo, int lHo, int lDo,
    int Cout, int stride, int pad, int Ktot, int ngroups) {
  __shared__ __align__(16) _Float16 As[32 * 48];       // [m][k], stride 48
#if CDNA5_HAVE_TDM
  __shared__ __align__(16) _Float16 Bs[4][2][32 * 32]; // [wave][buf][k][n]
#endif

  const int tid  = threadIdx.x;
  const int lane = tid & 31;
  const int wave = __builtin_amdgcn_readfirstlane(tid >> 5);
  const int m0   = (blockIdx.x / ngroups) * 32;
  const int msel  = wave & 1;                   // which 16-row half of A tile
  const int npair = (blockIdx.x % ngroups) * 2 + (wave >> 1);
  const bool active = npair < (Cout >> 5);
  const int n0 = npair * 32;

  // ---- A staging role: thread owns k-column kk for rows mr0 + 4*i, i=0..7
  const int kk  = tid & 31;
  const int mr0 = tid >> 5;
  int dib[8], hib[8], wib[8], nbb[8];
#pragma unroll
  for (int i = 0; i < 8; ++i) {
    int m  = m0 + mr0 + i * 4;
    int wo = m & ((1 << lWo) - 1);
    int ho = (m >> lWo) & ((1 << lHo) - 1);
    int dd = (m >> (lWo + lHo)) & ((1 << lDo) - 1);
    int nb = m >> (lWo + lHo + lDo);
    dib[i] = dd * stride - pad;
    hib[i] = ho * stride - pad;
    wib[i] = wo * stride - pad;
    nbb[i] = nb;
  }

  // WMMA 16-bit A/B fragment k-offsets (ISA 7.12.2)
  const int halfw = lane >> 4;
  const int row   = lane & 15;
  const int offK0 = halfw * 8;
  const int offK1 = 16 + halfw * 8;

  float8 acc0 = {0.f,0.f,0.f,0.f,0.f,0.f,0.f,0.f};
  float8 acc1 = {0.f,0.f,0.f,0.f,0.f,0.f,0.f,0.f};

  const int nsteps = (Ktot + 31) >> 5;
#if CDNA5_HAVE_TDM
  const unsigned bs0 = (unsigned)(unsigned long long)&Bs[wave][0][0];
  if (active)  // prologue: DMA B tile for step 0 into buffer 0
    tdm_load_f16_tile(bs0, wgt + (size_t)n0, Cout - n0, Ktot, 0,
                      32, 32, 0, Cout, 0);
#else
  const _Float16* wcol0 = wgt + n0 + row;
  const _Float16* wcol1 = wgt + n0 + 16 + row;
#endif

  for (int s = 0; s < nsteps; ++s) {
    const int k0 = s << 5;
#if CDNA5_HAVE_TDM
    if (active && (s + 1) < nsteps)  // DMA next B tile into the other buffer
      tdm_load_f16_tile(bs0 + (unsigned)(((s + 1) & 1) * 32 * 32 * 2),
                        wgt + (size_t)(k0 + 32) * Cout + n0,
                        Cout - n0, Ktot - (k0 + 32), 0, 32, 32, 0, Cout, 0);
#endif
    // ---- A gather -> LDS (all 128 threads), constant-divisor k decompose
    int k = k0 + kk;
    int ci, r2;
    if (STEM) { ci = k % 3; r2 = k / 3; }
    else      { ci = k & ((1 << cshift) - 1); r2 = k >> cshift; }
    int kw = r2 % KK; r2 /= KK;
    int kh = r2 % KK;
    int kd = r2 / KK;
    const bool kval = k < Ktot;
    __syncthreads();
#pragma unroll
    for (int i = 0; i < 8; ++i) {
      _Float16 v = (_Float16)0.f;
      if (kval) {
        int di = dib[i] + kd, hh = hib[i] + kh, wwi = wib[i] + kw;
        if ((unsigned)di < (unsigned)Di && (unsigned)hh < (unsigned)Hi &&
            (unsigned)wwi < (unsigned)Wi) {
          size_t idx = ((((size_t)nbb[i] * Di + di) * Hi + hh) * Wi + wwi) *
                           (size_t)Cin + ci;
          v = act[idx];
        }
      }
      As[(mr0 + i * 4) * 48 + kk] = v;
    }
    __syncthreads();

    if (active) {
      const int ar = msel * 16 + row;
      half8 alo = *(const half8*)&As[ar * 48 + offK0];
      half8 ahi = *(const half8*)&As[ar * 48 + offK1];
      half16 a = hcat(alo, ahi);
      half16 b0, b1;
#if CDNA5_HAVE_TDM
      // s_wait_tensorcnt takes an immediate: keep 1 in flight while the
      // next tile's DMA is outstanding, drain fully on the last step.
      if ((s + 1) < nsteps) __builtin_amdgcn_s_wait_tensorcnt(1);
      else                  __builtin_amdgcn_s_wait_tensorcnt(0);
      const _Float16* bb = &Bs[wave][s & 1][0];
#pragma unroll
      for (int e = 0; e < 8; ++e) {
        int kb0 = (offK0 + e) * 32, kb1 = (offK1 + e) * 32;
        b0[e]     = bb[kb0 + row];      b0[e + 8] = bb[kb1 + row];
        b1[e]     = bb[kb0 + 16 + row]; b1[e + 8] = bb[kb1 + 16 + row];
      }
#else
#pragma unroll
      for (int e = 0; e < 8; ++e) {
        int kb0 = k0 + offK0 + e, kb1 = k0 + offK1 + e;
        b0[e]     = (kb0 < Ktot) ? wcol0[(size_t)kb0 * Cout] : (_Float16)0.f;
        b0[e + 8] = (kb1 < Ktot) ? wcol0[(size_t)kb1 * Cout] : (_Float16)0.f;
        b1[e]     = (kb0 < Ktot) ? wcol1[(size_t)kb0 * Cout] : (_Float16)0.f;
        b1[e + 8] = (kb1 < Ktot) ? wcol1[(size_t)kb1 * Cout] : (_Float16)0.f;
      }
#endif
      acc0 = wmma16x16x32(a, b0, acc0);
      acc1 = wmma16x16x32(a, b1, acc1);
    }
  }

  if (active) {
    const int mB = m0 + msel * 16;
#pragma unroll
    for (int r = 0; r < 8; ++r) {  // C/D layout: VGPR r -> M=r (+8 hi half)
      int m = mB + r + halfw * 8;
      out[(size_t)m * Cout + n0 + row]      = acc0[r];
      out[(size_t)m * Cout + n0 + 16 + row] = acc1[r];
    }
  }
}

// ---------------------------------------------------------------------------
// Transpose-conv 2x2x2 stride 2 (VALID): out parity (a,b,c) uses tap
// (1-a,1-b,1-c) -> 8 parity GEMMs sharing one A fragment (8 WMMAs/K-step).
// All 8 tap tiles fetched per K-step by ONE 3-D TDM load (z = tap,
// dim1_stride = Cin*Cout). A rows are channel-contiguous -> direct b128.
// ---------------------------------------------------------------------------
__global__ __launch_bounds__(128) void deconv2x2x2_wmma_kernel(
    const _Float16* __restrict__ act, const _Float16* __restrict__ wgt,
    float* __restrict__ out,
    int lDi, int lHi, int lWi, int Cin, int Cout, int total_tiles) {
#if CDNA5_HAVE_TDM
  __shared__ __align__(16) _Float16 Bs[4][8 * 32 * 16];  // [wave][tap][k][n]
#endif
  const int wave = __builtin_amdgcn_readfirstlane(threadIdx.x >> 5);
  int tile = blockIdx.x * 4 + wave;
  if (tile >= total_tiles) return;
  const int n_tiles = Cout >> 4;
  const int m0 = (tile / n_tiles) * 16;
  const int n0 = (tile % n_tiles) * 16;
  const int lane  = threadIdx.x & 31;
  const int halfw = lane >> 4;
  const int row   = lane & 15;
  const int offK0 = halfw * 8, offK1 = 16 + halfw * 8;

  const float8 fzero = {0.f,0.f,0.f,0.f,0.f,0.f,0.f,0.f};
  float8 acc[8];
#pragma unroll
  for (int p = 0; p < 8; ++p) acc[p] = fzero;

  const _Float16* arow = act + (size_t)(m0 + row) * Cin;
#if CDNA5_HAVE_TDM
  _Float16* bsw = &Bs[wave][0];
  const unsigned bs_off = (unsigned)(unsigned long long)bsw;
#endif

  for (int k0 = 0; k0 < Cin; k0 += 32) {  // Cin is a multiple of 32
#if CDNA5_HAVE_TDM
    tdm_load_f16_tile(bs_off, wgt + (size_t)k0 * Cout + n0,
                      Cout - n0, Cin - k0, 8, 16, 32, 8,
                      Cout, (long long)Cin * Cout);
#endif
    half8 alo = *(const half8*)(arow + k0 + offK0);
    half8 ahi = *(const half8*)(arow + k0 + offK1);
    half16 a = hcat(alo, ahi);
#if CDNA5_HAVE_TDM
    __builtin_amdgcn_s_wait_tensorcnt(0);
#endif
#pragma unroll
    for (int p = 0; p < 8; ++p) {
      half16 b;
#if CDNA5_HAVE_TDM
      const _Float16* bb = bsw + (7 - p) * (32 * 16);  // tap = 1 - parity
#pragma unroll
      for (int e = 0; e < 8; ++e) {
        b[e]     = bb[(offK0 + e) * 16 + row];
        b[e + 8] = bb[(offK1 + e) * 16 + row];
      }
#else
      const _Float16* wcol = wgt + (size_t)(7 - p) * Cin * Cout + n0 + row;
#pragma unroll
      for (int e = 0; e < 8; ++e) {
        b[e]     = wcol[(size_t)(k0 + offK0 + e) * Cout];
        b[e + 8] = wcol[(size_t)(k0 + offK1 + e) * Cout];
      }
#endif
      acc[p] = wmma16x16x32(a, b, acc[p]);
    }
  }

  const int Wi = 1 << lWi, Hi = 1 << lHi, Di = 1 << lDi;
  const int Ho = 2 * Hi, Wo = 2 * Wi, Do = 2 * Di;
#pragma unroll
  for (int r = 0; r < 8; ++r) {
    int m  = m0 + r + halfw * 8;
    int wi = m & (Wi - 1);
    int hi = (m >> lWi) & (Hi - 1);
    int di = (m >> (lWi + lHi)) & (Di - 1);
    int nb = m >> (lWi + lHi + lDi);
#pragma unroll
    for (int p = 0; p < 8; ++p) {
      int od = 2 * di + (p >> 2);
      int oh = 2 * hi + ((p >> 1) & 1);
      int ow = 2 * wi + (p & 1);
      size_t o = ((((size_t)nb * Do + od) * Ho + oh) * Wo + ow) * (size_t)Cout +
                 n0 + row;
      out[o] = acc[p][r];
    }
  }
}

// ---------------------------------------------------------------------------
// Training-mode BN: per-channel sum & sumsq (one block per channel)
// ---------------------------------------------------------------------------
__global__ __launch_bounds__(256) void bn_stats_kernel(
    const float* __restrict__ x, float* __restrict__ stats, int M, int C) {
  const int c = blockIdx.x;
  float s = 0.f, s2 = 0.f;
  for (int m = threadIdx.x; m < M; m += 256) {
    float v = x[(size_t)m * C + c];
    s += v; s2 += v * v;
  }
  __shared__ float sh[512];
  sh[threadIdx.x] = s;
  sh[256 + threadIdx.x] = s2;
  __syncthreads();
  for (int st = 128; st > 0; st >>= 1) {
    if (threadIdx.x < st) {
      sh[threadIdx.x] += sh[threadIdx.x + st];
      sh[256 + threadIdx.x] += sh[256 + threadIdx.x + st];
    }
    __syncthreads();
  }
  if (threadIdx.x == 0) {
    stats[c] = sh[0];
    stats[C + c] = sh[256];
  }
}

// ---------------------------------------------------------------------------
// BN normalize + optional residual + optional ReLU; emits f32 and/or f16.
// C is always a power of two -> channel index via mask.
// ---------------------------------------------------------------------------
__global__ __launch_bounds__(256) void bn_apply_kernel(
    const float* __restrict__ x, const float* __restrict__ stats,
    const float* __restrict__ g, const float* __restrict__ b,
    const float* __restrict__ resid, float* __restrict__ out32,
    _Float16* __restrict__ out16, int M, int C, int do_relu) {
  const long total = (long)M * C;
  const float invM = 1.0f / (float)M;
  const long cmask = (long)(C - 1);
  for (long i = (long)blockIdx.x * 256 + threadIdx.x; i < total;
       i += (long)gridDim.x * 256) {
    int c = (int)(i & cmask);
    float mean = stats[c] * invM;
    float var  = stats[C + c] * invM - mean * mean;
    float y = (x[i] - mean) * rsqrtf(var + 1e-5f) * g[c] + b[c];
    if (resid) y += resid[i];
    if (do_relu) y = fmaxf(y, 0.f);
    if (out32) out32[i] = y;
    if (out16) out16[i] = (_Float16)y;
  }
}

// ---------------------------------------------------------------------------
// Host orchestration
// ---------------------------------------------------------------------------
namespace {
struct Blk { const float *c1, *g1, *b1, *c2, *g2, *b2, *dw, *dg, *db; };
struct Net {
  const float* x;
  const float *conv1_w, *bn1_g, *bn1_b;
  Blk L[4];
  const float *up_w[3], *up_g[3], *up_b[3];
  const float* final_w;
};
static int ilog2(int v) { int l = 0; while ((1 << l) < v) ++l; return l; }
}  // namespace

extern "C" void kernel_launch(void* const* d_in, const int* in_sizes, int n_in,
                              void* d_out, int out_size, void* d_ws,
                              size_t ws_size, hipStream_t stream) {
  auto IN = [&](int i) { return (const float*)d_in[i]; };

  // ---- input order detection: insertion order vs pytree(sorted-keys) order
  Net P{};
  const int X_ELEMS = 2 * 64 * 64 * 64 * 3;      // 1572864
  const int CONV1W_ELEMS = 7 * 7 * 7 * 3 * 64;   // 65856
  auto fill_sorted = [&](int base) {
    P.bn1_b = IN(base + 0); P.bn1_g = IN(base + 1);
    P.conv1_w = IN(base + 2); P.final_w = IN(base + 3);
    int o = base + 4;
    for (int s = 0; s < 4; ++s) {
      P.L[s].b1 = IN(o); P.L[s].g1 = IN(o + 1);
      P.L[s].b2 = IN(o + 2); P.L[s].g2 = IN(o + 3);
      P.L[s].c1 = IN(o + 4); P.L[s].c2 = IN(o + 5); o += 6;
      if (s > 0) { P.L[s].db = IN(o); P.L[s].dg = IN(o + 1); P.L[s].dw = IN(o + 2); o += 3; }
    }
    for (int j = 0; j < 3; ++j) {
      P.up_b[j] = IN(o); P.up_g[j] = IN(o + 1); P.up_w[j] = IN(o + 2); o += 3;
    }
  };
  auto fill_insert = [&]() {
    P.x = IN(0);
    P.conv1_w = IN(1); P.bn1_g = IN(2); P.bn1_b = IN(3);
    int o = 4;
    for (int s = 0; s < 4; ++s) {
      P.L[s].c1 = IN(o); P.L[s].g1 = IN(o + 1); P.L[s].b1 = IN(o + 2);
      P.L[s].c2 = IN(o + 3); P.L[s].g2 = IN(o + 4); P.L[s].b2 = IN(o + 5); o += 6;
      if (s > 0) { P.L[s].dw = IN(o); P.L[s].dg = IN(o + 1); P.L[s].db = IN(o + 2); o += 3; }
    }
    for (int j = 0; j < 3; ++j) {
      P.up_w[j] = IN(o); P.up_g[j] = IN(o + 1); P.up_b[j] = IN(o + 2); o += 3;
    }
    P.final_w = IN(o);
  };
  if (n_in > 0 && in_sizes[0] == X_ELEMS) {
    if (n_in > 1 && in_sizes[1] == CONV1W_ELEMS) fill_insert();
    else { P.x = IN(0); fill_sorted(1); }
  } else {
    fill_sorted(0);
    P.x = IN(n_in - 1);
  }

  // ---- workspace carve-up
  char* ws = (char*)d_ws;
  _Float16* F16A = (_Float16*)(ws + ((size_t)0 << 20));
  _Float16* F16B = (_Float16*)(ws + ((size_t)16 << 20));
  _Float16* WF16 = (_Float16*)(ws + ((size_t)32 << 20));
  float* ACT32 = (float*)(ws + ((size_t)48 << 20));
  float* CONV  = (float*)(ws + ((size_t)72 << 20));
  float* SC    = (float*)(ws + ((size_t)96 << 20));
  float* STATS = (float*)(ws + ((size_t)120 << 20));
  (void)ws_size; (void)out_size;

  auto cvt = [&](const float* s, _Float16* d, int n) {
    f32_to_f16_kernel<<<(n + 255) / 256, 256, 0, stream>>>(s, d, n);
  };
  auto stats = [&](const float* xx, int M, int C) {
    bn_stats_kernel<<<C, 256, 0, stream>>>(xx, STATS, M, C);
  };
  auto apply = [&](const float* xx, const float* g, const float* b,
                   const float* resid, float* o32, _Float16* o16, int M, int C,
                   int relu) {
    long total = (long)M * C;
    bn_apply_kernel<<<(int)((total + 255) / 256), 256, 0, stream>>>(
        xx, STATS, g, b, resid, o32, o16, M, C, relu);
  };
  auto convK = [&](auto kern, const _Float16* a, const float* w32, float* o,
                   int Din, int Cin, int Dout, int Cout, int Kk, int st,
                   int pad) {
    int Ktot = Kk * Kk * Kk * Cin;
    cvt(w32, WF16, Ktot * Cout);
    int M = 2 * Dout * Dout * Dout;
    int ngroups = ((Cout / 32) + 1) / 2;
    int lo = ilog2(Dout);
    kern<<<(M / 32) * ngroups, 128, 0, stream>>>(
        a, WF16, o, Din, Din, Din, Cin, ilog2(Cin), lo, lo, lo, Cout, st, pad,
        Ktot, ngroups);
  };
  auto deconv = [&](const _Float16* a, const float* w32, float* o, int Din,
                    int Cin, int Cout) {
    cvt(w32, WF16, 8 * Cin * Cout);
    int M = 2 * Din * Din * Din;
    int tiles = (M / 16) * (Cout / 16);
    int ld = ilog2(Din);
    deconv2x2x2_wmma_kernel<<<(tiles + 3) / 4, 128, 0, stream>>>(
        a, WF16, o, ld, ld, ld, Cin, Cout, tiles);
  };

  // ---- stem: 7^3 s2 SAME (pad_lo 2), 3 -> 64 ch @ 32^3
  cvt(P.x, F16A, X_ELEMS);
  convK(conv3d_wmma_kernel<7, true>, F16A, P.conv1_w, CONV, 64, 3, 32, 64, 7, 2, 2);
  stats(CONV, 65536, 64);
  apply(CONV, P.bn1_g, P.bn1_b, nullptr, ACT32, F16A, 65536, 64, 1);

  // ---- residual stages
  const int DinS[4]  = {32, 32, 16, 8};
  const int DoutS[4] = {32, 16, 8, 4};
  const int CinS[4]  = {64, 64, 128, 256};
  const int CoutS[4] = {64, 128, 256, 512};
  const int StrS[4]  = {1, 2, 2, 2};
  for (int s = 0; s < 4; ++s) {
    const Blk& B = P.L[s];
    int Din = DinS[s], Dout = DoutS[s], Ci = CinS[s], Co = CoutS[s],
        st = StrS[s];
    int M = 2 * Dout * Dout * Dout;
    const float* scRes;
    if (B.dw) {  // projection shortcut: 1x1 s2 (pad 0) + BN, no relu
      convK(conv3d_wmma_kernel<1, false>, F16A, B.dw, CONV, Din, Ci, Dout, Co, 1, st, 0);
      stats(CONV, M, Co);
      apply(CONV, B.dg, B.db, nullptr, SC, nullptr, M, Co, 0);
      scRes = SC;
    } else {
      scRes = ACT32;
    }
    convK(conv3d_wmma_kernel<3, false>, F16A, B.c1, CONV, Din, Ci, Dout, Co, 3,
          st, st == 1 ? 1 : 0);  // SAME: pad_lo 1 (s1) / 0 (s2, even in)
    stats(CONV, M, Co);
    apply(CONV, B.g1, B.b1, nullptr, nullptr, F16B, M, Co, 1);
    convK(conv3d_wmma_kernel<3, false>, F16B, B.c2, CONV, Dout, Co, Dout, Co, 3, 1, 1);
    stats(CONV, M, Co);
    apply(CONV, B.g2, B.b2, scRes, ACT32, F16A, M, Co, 1);
  }

  // ---- decoder: 3x (transpose conv 2^3 s2 + BN + ReLU)
  const int dD[3]  = {4, 8, 16};
  const int dCi[3] = {512, 256, 128};
  const int dCo[3] = {256, 128, 64};
  for (int j = 0; j < 3; ++j) {
    deconv(F16A, P.up_w[j], CONV, dD[j], dCi[j], dCo[j]);
    int Mo = 2 * (2 * dD[j]) * (2 * dD[j]) * (2 * dD[j]);
    stats(CONV, Mo, dCo[j]);
    apply(CONV, P.up_g[j], P.up_b[j], nullptr, nullptr, F16A, Mo, dCo[j], 1);
  }

  // ---- head: 1x1x1, 64 -> 32, straight to d_out (f32)
  convK(conv3d_wmma_kernel<1, false>, F16A, P.final_w, (float*)d_out, 32, 64,
        32, 32, 1, 1, 0);
}
